// Attention_30631706755638
// MI455X (gfx1250) — matrix operand order repaired and checked by reading, not compiled, not run on hardware
//
#include <hip/hip_runtime.h>
#include <hip/hip_bf16.h>

// ---------------------------------------------------------------------------
// Causal attention, bf16 WMMA pipeline for gfx1250 (MI455X).
//   B=4, S=2048, D=768 (single-head, full width).
// GEMMs use 2x4 register blocking (32x64 per wave) + software-pipelined
// double-buffered fragment loads so WMMA overlaps global load latency.
// ---------------------------------------------------------------------------

#define B_N   4
#define S_LEN 2048
#define D_DIM 768

typedef __attribute__((ext_vector_type(16))) __bf16 v16bf;
typedef __attribute__((ext_vector_type(8)))  __bf16 v8bf;
typedef __attribute__((ext_vector_type(8)))  float  v8f;

#define WMMA_BF16(A, B, C) \
    __builtin_amdgcn_wmma_f32_16x16x32_bf16(false, (A), false, (B), (short)0, (C), false, false)

// f32 -> bf16, round-to-nearest-even
__device__ __forceinline__ unsigned short f2bf(float f) {
    unsigned u = __builtin_bit_cast(unsigned, f);
    unsigned r = u + 0x7FFFu + ((u >> 16) & 1u);
    return (unsigned short)(r >> 16);
}

// A-matrix 16x32 bf16 fragment (ISA 7.12.2): lanes 0-15 hold row m=lane with
// K = {0..7, 16..23}; lanes 16-31 hold row m=lane-16 with K = {8..15, 24..31}.
__device__ __forceinline__ v16bf load_a_frag(const __bf16* row, int k0, int hi) {
    union { v16bf f; v8bf h[2]; } u;
    const __bf16* p = row + k0 + (hi << 3);
    u.h[0] = *(const v8bf*)(p);
    u.h[1] = *(const v8bf*)(p + 16);
    return u.f;
}

// B-matrix 32x16 bf16 fragment: lanes 0-15 hold col n=lane, K={0..7,16..23};
// lanes 16-31 hold col n=lane-16, K={8..15,24..31}. Row pointer = W[n,:] (K-contig).
__device__ __forceinline__ v16bf load_b_frag(const __bf16* row, int k0, int hi) {
    union { v16bf f; v8bf h[2]; } u;
    const __bf16* p = row + k0 + (hi << 3);
    u.h[0] = *(const v8bf*)(p);
    u.h[1] = *(const v8bf*)(p + 16);
    return u.f;
}

// ---------------------------------------------------------------------------
// Wave-level 32x64 GEMM core, K-loop with register double buffering.
// acc[0..3]: rows [0,16) x col tiles 0..3 ; acc[4..7]: rows [16,32).
// ---------------------------------------------------------------------------
__device__ __forceinline__ void wave_mma_2x4(
    const __bf16* __restrict__ arow0, const __bf16* __restrict__ arow1,
    const __bf16* __restrict__ br0, const __bf16* __restrict__ br1,
    const __bf16* __restrict__ br2, const __bf16* __restrict__ br3,
    int K, int hi, v8f acc[8]) {

    v16bf a0 = load_a_frag(arow0, 0, hi);
    v16bf a1 = load_a_frag(arow1, 0, hi);
    v16bf b0 = load_b_frag(br0, 0, hi);
    v16bf b1 = load_b_frag(br1, 0, hi);
    v16bf b2 = load_b_frag(br2, 0, hi);
    v16bf b3 = load_b_frag(br3, 0, hi);

    for (int k0 = 0; k0 < K; k0 += 32) {
        // issue next iteration's fragment loads first (stay in flight during WMMA)
        const int kn = (k0 + 32 < K) ? (k0 + 32) : k0;
        __builtin_prefetch(arow0 + k0 + 256, 0, 1);
        __builtin_prefetch(arow1 + k0 + 256, 0, 1);
        v16bf na0 = load_a_frag(arow0, kn, hi);
        v16bf na1 = load_a_frag(arow1, kn, hi);
        v16bf nb0 = load_b_frag(br0, kn, hi);
        v16bf nb1 = load_b_frag(br1, kn, hi);
        v16bf nb2 = load_b_frag(br2, kn, hi);
        v16bf nb3 = load_b_frag(br3, kn, hi);

        acc[0] = WMMA_BF16(a0, b0, acc[0]);
        acc[1] = WMMA_BF16(a0, b1, acc[1]);
        acc[2] = WMMA_BF16(a0, b2, acc[2]);
        acc[3] = WMMA_BF16(a0, b3, acc[3]);
        acc[4] = WMMA_BF16(a1, b0, acc[4]);
        acc[5] = WMMA_BF16(a1, b1, acc[5]);
        acc[6] = WMMA_BF16(a1, b2, acc[6]);
        acc[7] = WMMA_BF16(a1, b3, acc[7]);

        a0 = na0; a1 = na1;
        b0 = nb0; b1 = nb1; b2 = nb2; b3 = nb3;
    }
}

// ---------------------------------------------------------------------------
// Kernel 1: f32 -> bf16 conversion (n must be a multiple of 4)
// ---------------------------------------------------------------------------
__global__ __launch_bounds__(256)
void attn_cvt_bf16(const float* __restrict__ src, unsigned short* __restrict__ dst, int n) {
    int i = (blockIdx.x * 256 + threadIdx.x) * 4;
    if (i >= n) return;
    float4 f = *(const float4*)(src + i);
    ushort4 o;
    o.x = f2bf(f.x); o.y = f2bf(f.y); o.z = f2bf(f.z); o.w = f2bf(f.w);
    *(ushort4*)(dst + i) = o;
}

// ---------------------------------------------------------------------------
// Kernel 2: QKV projection.  y = x @ W^T.  Block tile 128x64, 4 waves,
// wave owns rows [wave*32, wave*32+32).  blockIdx.z selects Wq/Wk/Wv.
// V output is stored transposed: vt[b][d][s].
// ---------------------------------------------------------------------------
__global__ __launch_bounds__(128)
void attn_qkv_gemm(const __bf16* __restrict__ xb, const __bf16* __restrict__ w_all,
                   unsigned short* __restrict__ qb, unsigned short* __restrict__ kb,
                   unsigned short* __restrict__ vt) {
    const int m0    = blockIdx.x * 128;
    const int n0    = blockIdx.y * 64;
    const int which = blockIdx.z;
    const __bf16* W = w_all + (size_t)which * (D_DIM * D_DIM);

    const int lane = threadIdx.x & 31;
    const int wave = threadIdx.x >> 5;
    const int lr   = lane & 15;
    const int hi   = lane >> 4;

    const __bf16* arow0 = xb + (size_t)(m0 + wave * 32 + lr) * D_DIM;
    const __bf16* arow1 = arow0 + (size_t)16 * D_DIM;
    const __bf16* br0 = W + (size_t)(n0 +  0 + lr) * D_DIM;
    const __bf16* br1 = W + (size_t)(n0 + 16 + lr) * D_DIM;
    const __bf16* br2 = W + (size_t)(n0 + 32 + lr) * D_DIM;
    const __bf16* br3 = W + (size_t)(n0 + 48 + lr) * D_DIM;

    v8f acc[8] = {};
    wave_mma_2x4(arow0, arow1, br0, br1, br2, br3, D_DIM, hi, acc);

    const int mrow = m0 + wave * 32 + hi * 8;
#pragma unroll
    for (int g = 0; g < 2; ++g) {        // two 16-row tiles
#pragma unroll
        for (int t = 0; t < 4; ++t) {
            const int col = n0 + t * 16 + lr;
#pragma unroll
            for (int r = 0; r < 8; ++r) {
                const int row = mrow + g * 16 + r;
                const unsigned short v = f2bf(acc[g * 4 + t][r]);
                if (which == 0) {
                    qb[(size_t)row * D_DIM + col] = v;
                } else if (which == 1) {
                    kb[(size_t)row * D_DIM + col] = v;
                } else {
                    const int b = row >> 11, s = row & (S_LEN - 1);
                    vt[((size_t)b * D_DIM + col) * S_LEN + s] = v;
                }
            }
        }
    }
}

// ---------------------------------------------------------------------------
// Kernel 3: scores = (q @ k^T) * 1/sqrt(768), f32 output.  Block tile:
// 128 query rows x 64 key cols.  Tiles fully above the diagonal exit early.
// ---------------------------------------------------------------------------
__global__ __launch_bounds__(128)
void attn_scores_gemm(const __bf16* __restrict__ qb, const __bf16* __restrict__ kb,
                      float* __restrict__ sc) {
    const int n0 = blockIdx.x * 64;    // key columns
    const int m0 = blockIdx.y * 128;   // query rows
    const int b  = blockIdx.z;
    if (n0 > m0 + 127) return;         // fully masked tile

    const int lane = threadIdx.x & 31;
    const int wave = threadIdx.x >> 5;
    const int lr   = lane & 15;
    const int hi   = lane >> 4;

    const __bf16* qbase = qb + (size_t)b * S_LEN * D_DIM;
    const __bf16* kbase = kb + (size_t)b * S_LEN * D_DIM;

    const __bf16* arow0 = qbase + (size_t)(m0 + wave * 32 + lr) * D_DIM;
    const __bf16* arow1 = arow0 + (size_t)16 * D_DIM;
    const __bf16* br0 = kbase + (size_t)(n0 +  0 + lr) * D_DIM;
    const __bf16* br1 = kbase + (size_t)(n0 + 16 + lr) * D_DIM;
    const __bf16* br2 = kbase + (size_t)(n0 + 32 + lr) * D_DIM;
    const __bf16* br3 = kbase + (size_t)(n0 + 48 + lr) * D_DIM;

    v8f acc[8] = {};
    wave_mma_2x4(arow0, arow1, br0, br1, br2, br3, D_DIM, hi, acc);

    const float scale = 0.036084391824351615f;  // 1/sqrt(768)
    const int   mrow  = m0 + wave * 32 + hi * 8;
    float* out = sc + (size_t)b * S_LEN * S_LEN;
#pragma unroll
    for (int g = 0; g < 2; ++g) {
#pragma unroll
        for (int t = 0; t < 4; ++t) {
            const int col = n0 + t * 16 + lr;
#pragma unroll
            for (int r = 0; r < 8; ++r)
                out[(size_t)(mrow + g * 16 + r) * S_LEN + col] = acc[g * 4 + t][r] * scale;
        }
    }
}

// ---------------------------------------------------------------------------
// Kernel 4: causal row softmax -> bf16 probabilities, masked region zeroed
// so the P @ V GEMM can be dense.  One wave32 per row.
// ---------------------------------------------------------------------------
__global__ __launch_bounds__(256)
void attn_softmax(const float* __restrict__ sc, unsigned short* __restrict__ pb) {
    const int lane = threadIdx.x & 31;
    const int wave = threadIdx.x >> 5;
    const int q    = blockIdx.x * 8 + wave;
    const int b    = blockIdx.y;

    const float*    row  = sc + ((size_t)b * S_LEN + q) * S_LEN;
    unsigned short* prow = pb + ((size_t)b * S_LEN + q) * S_LEN;
    const int nv = q + 1;  // valid (unmasked) length

    float mx = -3.0e38f;
    for (int j = lane; j < nv; j += 32) mx = fmaxf(mx, row[j]);
#pragma unroll
    for (int off = 16; off; off >>= 1) mx = fmaxf(mx, __shfl_xor(mx, off, 32));

    float sum = 0.0f;
    for (int j = lane; j < nv; j += 32) sum += __expf(row[j] - mx);
#pragma unroll
    for (int off = 16; off; off >>= 1) sum += __shfl_xor(sum, off, 32);

    const float inv = 1.0f / sum;
    for (int j = lane; j < S_LEN; j += 32) {
        unsigned short v = 0;
        if (j < nv) v = f2bf(__expf(row[j] - mx) * inv);
        prow[j] = v;
    }
}

// ---------------------------------------------------------------------------
// Kernel 5: out = P @ V.   A = P[b] rows (K contiguous, K = 2048),
// B fragments come from vt[b] (V pre-transposed so vt row d is K-contiguous).
// ---------------------------------------------------------------------------
__global__ __launch_bounds__(128)
void attn_pv_gemm(const __bf16* __restrict__ pb, const __bf16* __restrict__ vt,
                  float* __restrict__ out) {
    const int m0 = blockIdx.x * 128;  // query rows within batch
    const int n0 = blockIdx.y * 64;   // feature columns
    const int b  = blockIdx.z;

    const int lane = threadIdx.x & 31;
    const int wave = threadIdx.x >> 5;
    const int lr   = lane & 15;
    const int hi   = lane >> 4;

    const __bf16* pbase = pb + (size_t)b * S_LEN * S_LEN;
    const __bf16* vbase = vt + (size_t)b * D_DIM * S_LEN;

    const __bf16* arow0 = pbase + (size_t)(m0 + wave * 32 + lr) * S_LEN;
    const __bf16* arow1 = arow0 + (size_t)16 * S_LEN;
    const __bf16* br0 = vbase + (size_t)(n0 +  0 + lr) * S_LEN;
    const __bf16* br1 = vbase + (size_t)(n0 + 16 + lr) * S_LEN;
    const __bf16* br2 = vbase + (size_t)(n0 + 32 + lr) * S_LEN;
    const __bf16* br3 = vbase + (size_t)(n0 + 48 + lr) * S_LEN;

    v8f acc[8] = {};
    wave_mma_2x4(arow0, arow1, br0, br1, br2, br3, S_LEN, hi, acc);

    const int mrow = m0 + wave * 32 + hi * 8;
    float* obase = out + (size_t)b * S_LEN * D_DIM;
#pragma unroll
    for (int g = 0; g < 2; ++g) {
#pragma unroll
        for (int t = 0; t < 4; ++t) {
            const int col = n0 + t * 16 + lr;
#pragma unroll
            for (int r = 0; r < 8; ++r)
                obase[(size_t)(mrow + g * 16 + r) * D_DIM + col] = acc[g * 4 + t][r];
        }
    }
}

// ---------------------------------------------------------------------------
// Host launcher
// ---------------------------------------------------------------------------
extern "C" void kernel_launch(void* const* d_in, const int* in_sizes, int n_in,
                              void* d_out, int out_size, void* d_ws, size_t ws_size,
                              hipStream_t stream) {
    const float* x  = (const float*)d_in[0];
    const float* Wq = (const float*)d_in[1];
    const float* Wk = (const float*)d_in[2];
    const float* Wv = (const float*)d_in[3];

    const size_t n_x = (size_t)B_N * S_LEN * D_DIM;   // 6,291,456
    const size_t n_w = (size_t)D_DIM * D_DIM;         //   589,824
    const size_t n_p = (size_t)B_N * S_LEN * S_LEN;   // 16,777,216

    // Workspace layout (256B aligned slabs)
    char* ws = (char*)d_ws;
    size_t off = 0;
    auto slab = [&](size_t bytes) -> void* {
        void* p = ws + off;
        off = (off + bytes + 255) & ~(size_t)255;
        return p;
    };
    unsigned short* xb    = (unsigned short*)slab(n_x * 2);
    unsigned short* w_all = (unsigned short*)slab(3 * n_w * 2);
    unsigned short* qb    = (unsigned short*)slab(n_x * 2);
    unsigned short* kb    = (unsigned short*)slab(n_x * 2);
    unsigned short* vt    = (unsigned short*)slab(n_x * 2);
    unsigned short* pbuf  = (unsigned short*)slab(n_p * 2);
    float*          sc    = (float*)slab(n_p * 4);

    // 1) convert inputs to bf16
    attn_cvt_bf16<<<dim3((unsigned)(n_x / 4 / 256)), 256, 0, stream>>>(x, xb, (int)n_x);
    attn_cvt_bf16<<<dim3((unsigned)(n_w / 4 / 256)), 256, 0, stream>>>(Wq, w_all, (int)n_w);
    attn_cvt_bf16<<<dim3((unsigned)(n_w / 4 / 256)), 256, 0, stream>>>(Wk, w_all + n_w, (int)n_w);
    attn_cvt_bf16<<<dim3((unsigned)(n_w / 4 / 256)), 256, 0, stream>>>(Wv, w_all + 2 * n_w, (int)n_w);

    // 2) QKV projections (z selects Wq/Wk/Wv; V stored transposed)
    attn_qkv_gemm<<<dim3(B_N * S_LEN / 128, D_DIM / 64, 3), 128, 0, stream>>>(
        (const __bf16*)xb, (const __bf16*)w_all, qb, kb, vt);

    // 3) causal-skipped score GEMM
    attn_scores_gemm<<<dim3(S_LEN / 64, S_LEN / 128, B_N), 128, 0, stream>>>(
        (const __bf16*)qb, (const __bf16*)kb, sc);

    // 4) softmax -> bf16 P (masked region zeroed)
    attn_softmax<<<dim3(S_LEN / 8, B_N), 256, 0, stream>>>(sc, pbuf);

    // 5) out = P @ V
    attn_pv_gemm<<<dim3(S_LEN / 128, D_DIM / 64, B_N), 128, 0, stream>>>(
        (const __bf16*)pbuf, (const __bf16*)vt, (float*)d_out);
}
